// LearnableLogisticModel_30210799960825
// MI455X (gfx1250) — matrix operand order repaired and checked by reading, not compiled, not run on hardware
//
#include <hip/hip_runtime.h>
#include <hip/hip_bf16.h>
#include <math.h>

typedef __attribute__((ext_vector_type(16))) _Float16 v16h;
typedef __attribute__((ext_vector_type(8)))  float    v8f;

#define DD 64
#define NSAMP 8192
#define WAVES_PER_BLOCK 4
#define ROWS_PER_BLOCK (WAVES_PER_BLOCK * 16)   // 64 samples per block
#define XS_LD 65                                 // LDS row pitch (bank-conflict pad)

// workspace layout (bytes)
#define WS_TT 0                                  // _Float16[4096*64] = 512 KB (Tt[col][i], gated)
#define WS_W2 (512 * 1024)                       // _Float16[64*64]   (W2 cols, gated)
#define WS_W1 (512 * 1024 + 8192)                // float[64]         (g1 * w1)
#define WS_S0 (512 * 1024 + 8192 + 256)          // float             (g0 * w0)

__device__ __forceinline__ int pair_idx(int i, int j) {     // i < j, triu_indices(64,1) order
    return i * 63 - (i * (i - 1)) / 2 + (j - i - 1);
}
__device__ __forceinline__ int tri_idx(int i, int j, int k) { // i < j < k, lex order
    int o1 = 41664 - ((64 - i) * (63 - i) * (62 - i)) / 6;
    int o2 = ((62 - i) * (63 - i)) / 2 - ((63 - j) * (64 - j)) / 2;
    return o1 + o2 + (k - j - 1);
}
__device__ __forceinline__ float gate(const float* alpha, int m) {
    float a0 = alpha[0], a1 = alpha[1], a2 = alpha[2], a3 = alpha[3];
    float mx = fmaxf(fmaxf(a0, a1), fmaxf(a2, a3));
    float e0 = expf(a0 - mx), e1 = expf(a1 - mx), e2 = expf(a2 - mx), e3 = expf(a3 - mx);
    float s = e0 + e1 + e2 + e3;
    float em = (m == 0) ? e0 : (m == 1) ? e1 : (m == 2) ? e2 : e3;
    return em / s;
}

// Build symmetric gated cubic tensor, stored transposed: tt[col=(j*64+k)][i], f16.
__global__ void build_tt_kernel(const float* __restrict__ w3,
                                const float* __restrict__ alpha,
                                _Float16* __restrict__ tt) {
    int t = blockIdx.x * blockDim.x + threadIdx.x;   // 0 .. 262143
    int i = t & 63;
    int col = t >> 6;
    int j = col >> 6, k = col & 63;
    int a = i, b = j, c = k, tmp;
    if (a > b) { tmp = a; a = b; b = tmp; }
    if (b > c) { tmp = b; b = c; c = tmp; }
    if (a > b) { tmp = a; a = b; b = tmp; }
    float v;
    if (a == c)      v = w3[a];                                        // x_a^3
    else if (a == b) v = w3[64 + pair_idx(a, c)] * (1.0f / 3.0f);      // x_a^2 x_c
    else if (b == c) v = w3[64 + 2016 + pair_idx(a, b)] * (1.0f / 3.0f); // x_a x_b^2
    else             v = w3[64 + 4032 + tri_idx(a, b, c)] * (1.0f / 6.0f);
    tt[t] = (_Float16)(v * gate(alpha, 3));
}

// Build gated symmetric W2 (col-major cols for B tiles), gated w1, gated w0.
__global__ void build_small_kernel(const float* __restrict__ w0,
                                   const float* __restrict__ w1,
                                   const float* __restrict__ w2,
                                   const float* __restrict__ alpha,
                                   _Float16* __restrict__ w2t,
                                   float* __restrict__ w1g,
                                   float* __restrict__ s0) {
    int t = blockIdx.x * blockDim.x + threadIdx.x;   // 0 .. 4095
    int k = t >> 6, i = t & 63;
    float v;
    if (i == k) v = w2[i];
    else {
        int a = (i < k) ? i : k, b = (i < k) ? k : i;
        v = 0.5f * w2[64 + pair_idx(a, b)];
    }
    w2t[t] = (_Float16)(v * gate(alpha, 2));
    if (t < 64) w1g[t] = w1[t] * gate(alpha, 1);
    if (t == 0) *s0 = w0[0] * gate(alpha, 0);
}

__global__ void __launch_bounds__(128, 1)
poly_main_kernel(const float* __restrict__ X,
                 const _Float16* __restrict__ tt,
                 const _Float16* __restrict__ w2t,
                 const float* __restrict__ w1g,
                 const float* __restrict__ s0p,
                 float* __restrict__ out) {
    __shared__ float Xs[ROWS_PER_BLOCK][XS_LD];
    __shared__ float w1s[64];
    __shared__ float fs1[ROWS_PER_BLOCK];
    __shared__ float fs3[ROWS_PER_BLOCK];

    const int tid = threadIdx.x;
    const int n0 = blockIdx.x * ROWS_PER_BLOCK;

    // Cooperative coalesced load of this block's 64x64 fp32 X tile into LDS.
    const float4* X4 = (const float4*)(X + (size_t)n0 * DD);
    for (int idx = tid; idx < ROWS_PER_BLOCK * 16; idx += 128) {
        int row = idx >> 4, c4 = idx & 15;
        float4 v = X4[row * 16 + c4];
        Xs[row][c4 * 4 + 0] = v.x; Xs[row][c4 * 4 + 1] = v.y;
        Xs[row][c4 * 4 + 2] = v.z; Xs[row][c4 * 4 + 3] = v.w;
    }
    if (tid < 64) w1s[tid] = w1g[tid];
    __syncthreads();

    const int lane = tid & 31;
    const int wv   = tid >> 5;
    const int nN   = lane & 15;   // WMMA column N of this lane
    const int kh   = lane >> 4;   // K half-wave group
    const int tb   = wv * 16;     // wave's 16-sample tile base row in Xs

    // ---- Pack A (16x64 f16) contiguous k = 16*kh + e per chunk, same order as B; kept in VGPRs all loop ----
    union V16 { v16h v; _Float16 h[16]; };
    V16 A0, A1;
    {
        const int row = tb + nN;  // lane's M = nN for A
        #pragma unroll
        for (int e = 0; e < 16; ++e) {
            int k0 = (kh << 4) + e;  // K within 16x32 chunk
            A0.h[e] = (_Float16)Xs[row][k0];
            A1.h[e] = (_Float16)Xs[row][k0 + 32];
        }
    }

    // ---- Pre-load epilogue multipliers into registers ----
    // D tile layout: acc[r] = D[M = r + 8*kh][N = nN];  k = 16*phase + nN
    float xkp[4][8];
    #pragma unroll
    for (int ph = 0; ph < 4; ++ph)
        #pragma unroll
        for (int r = 0; r < 8; ++r)
            xkp[ph][r] = Xs[tb + r + (kh << 3)][ph * 16 + nN];

    float accf[8];
    #pragma unroll
    for (int r = 0; r < 8; ++r) accf[r] = 0.0f;

    // ---- Cubic contraction: 256 column tiles (j fixed within each group of 4) ----
    for (int j = 0; j < 64; ++j) {
        int jn = (j + 1 < 64) ? (j + 1) : 63;
        __builtin_prefetch(tt + (size_t)jn * 4096 + lane * 128, 0, 3);

        float xj[8];
        #pragma unroll
        for (int r = 0; r < 8; ++r) xj[r] = Xs[tb + r + (kh << 3)][j];

        #pragma unroll
        for (int ph = 0; ph < 4; ++ph) {
            // B tile: 32x16 f16 per chunk; lane reads 16 contiguous halves (32B)
            const _Float16* bp = tt + ((size_t)(j * 64 + ph * 16 + nN) * 64 + kh * 16);
            v16h B0 = *(const v16h*)(bp);
            v16h B1 = *(const v16h*)(bp + 32);
            v8f acc = {};
            acc = __builtin_amdgcn_wmma_f32_16x16x32_f16(false, A0.v, false, B0,
                                                         (short)0, acc, false, false);
            acc = __builtin_amdgcn_wmma_f32_16x16x32_f16(false, A1.v, false, B1,
                                                         (short)0, acc, false, false);
            #pragma unroll
            for (int r = 0; r < 8; ++r)
                accf[r] = fmaf(acc[r], xj[r] * xkp[ph][r], accf[r]);
        }
    }

    // ---- Quadratic term: same pipeline, 4 extra B tiles from W2, multiplier x_k ----
    #pragma unroll
    for (int ph = 0; ph < 4; ++ph) {
        const _Float16* bp = w2t + ((size_t)(ph * 16 + nN) * 64 + kh * 16);
        v16h B0 = *(const v16h*)(bp);
        v16h B1 = *(const v16h*)(bp + 32);
        v8f acc = {};
        acc = __builtin_amdgcn_wmma_f32_16x16x32_f16(false, A0.v, false, B0,
                                                     (short)0, acc, false, false);
        acc = __builtin_amdgcn_wmma_f32_16x16x32_f16(false, A1.v, false, B1,
                                                     (short)0, acc, false, false);
        #pragma unroll
        for (int r = 0; r < 8; ++r)
            accf[r] = fmaf(acc[r], xkp[ph][r], accf[r]);
    }

    // ---- Linear term partial: lane covers sample (tb+nN), k-range [32*kh, 32*kh+32) ----
    float c1p = 0.0f;
    {
        const int row = tb + nN;
        #pragma unroll
        for (int k = 0; k < 32; ++k) {
            int kk = (kh << 5) + k;
            c1p = fmaf(Xs[row][kk], w1s[kk], c1p);
        }
    }
    float c1full = c1p + __shfl_xor(c1p, 16, 32);  // combine the two k-halves

    // ---- Deterministic cross-lane reduction of accf over the 16 N-lanes ----
    #pragma unroll
    for (int r = 0; r < 8; ++r) {
        float v = accf[r];
        v += __shfl_xor(v, 1, 16);
        v += __shfl_xor(v, 2, 16);
        v += __shfl_xor(v, 4, 16);
        v += __shfl_xor(v, 8, 16);
        accf[r] = v;
    }
    if (nN == 0) {
        #pragma unroll
        for (int r = 0; r < 8; ++r) fs3[tb + (kh << 3) + r] = accf[r];  // unique writer
    }
    if (kh == 0) fs1[tb + nN] = c1full;                                  // unique writer
    __syncthreads();

    const float s0 = *s0p;
    for (int l = tid; l < ROWS_PER_BLOCK; l += 128) {
        float f = fs3[l] + fs1[l] + s0;
        out[n0 + l] = 1.0f / (1.0f + expf(-f));
    }
}

extern "C" void kernel_launch(void* const* d_in, const int* in_sizes, int n_in,
                              void* d_out, int out_size, void* d_ws, size_t ws_size,
                              hipStream_t stream) {
    const float* X     = (const float*)d_in[0];
    const float* w0    = (const float*)d_in[1];
    const float* w1    = (const float*)d_in[2];
    const float* w2    = (const float*)d_in[3];
    const float* w3    = (const float*)d_in[4];
    const float* alpha = (const float*)d_in[5];

    char* ws = (char*)d_ws;
    _Float16* tt  = (_Float16*)(ws + WS_TT);
    _Float16* w2t = (_Float16*)(ws + WS_W2);
    float*    w1g = (float*)(ws + WS_W1);
    float*    s0  = (float*)(ws + WS_S0);
    float*    out = (float*)d_out;

    build_tt_kernel<<<dim3(1024), dim3(256), 0, stream>>>(w3, alpha, tt);
    build_small_kernel<<<dim3(16), dim3(256), 0, stream>>>(w0, w1, w2, alpha, w2t, w1g, s0);
    poly_main_kernel<<<dim3(NSAMP / ROWS_PER_BLOCK), dim3(128), 0, stream>>>(
        X, tt, w2t, w1g, s0, out);
}
